// MemTransformerLM_34677565948120
// MI455X (gfx1250) — matrix operand order repaired and checked by reading, not compiled, not run on hardware
//
#include <hip/hip_runtime.h>

// ---------------------------------------------------------------------------
// Transformer-XL relative attention layer for MI455X (gfx1250, wave32, WMMA).
// All GEMMs use v_wmma_f32_16x16x32_bf16 (bf16 in, f32 accumulate).
// GEMM A-tiles staged global->LDS via the Tensor Data Mover (TDM) when the
// toolchain exposes __builtin_amdgcn_tensor_load_to_lds; B-tiles staged
// cooperatively (TDM cannot transpose).  All shapes divide tiles exactly.
// ---------------------------------------------------------------------------

typedef __attribute__((ext_vector_type(16))) __bf16 v16bf;
typedef __attribute__((ext_vector_type(8)))  __bf16 v8bf;
typedef __attribute__((ext_vector_type(4)))  __bf16 v4bf;
typedef __attribute__((ext_vector_type(8)))  float  v8f;
typedef __attribute__((ext_vector_type(4)))  float  v4f;

#if defined(__gfx1250__) && __has_builtin(__builtin_amdgcn_tensor_load_to_lds)
#define USE_TDM 1
typedef __attribute__((ext_vector_type(4))) unsigned tdm_v4u;
typedef __attribute__((ext_vector_type(8))) int      tdm_v8i;
typedef __attribute__((ext_vector_type(4))) int      tdm_v4i;
#else
#define USE_TDM 0
#endif

#define QD    512
#define BD    8
#define NHD   16
#define DHD   64
#define DD    1024
#define MLEND 512
#define SPAND 512
#define KLEN  1024   // MLEN + Q
#define HD    1024   // NH*DH
#define HD3   3072   // 3*NH*DH

// ---------------------------------------------------------------------------
// WMMA helpers.  Fragment layouts per CDNA5 ISA 7.12.2 (wave32):
//  A 16x32 bf16: lane L<16 -> row L, K in {0..7,16..23}; lane L+16 -> K {8..15,24..31}
//  B 32x16 bf16: lane L<16 -> col L, K 0..15; lane L+16 -> K 16..31 (code's A/B order, not ISA)
//  C/D 16x16 f32: VGPR g, lanes 0-15 -> M=g, N=lane; lanes 16-31 -> M=g+8
// ---------------------------------------------------------------------------
__device__ __forceinline__ v8f wmma_bf16(v16bf a, v16bf b, v8f c) {
  return __builtin_amdgcn_wmma_f32_16x16x32_bf16(
      /*neg_a=*/false, a, /*neg_b=*/false, b,
      /*c_mod=*/(short)0, c, /*reuse_a=*/false, /*reuse_b=*/false);
}

// A operand from a row-major [rows, ld] buffer: per lane 2 contiguous 16B runs.
__device__ __forceinline__ v16bf load_a_frag(const __bf16* p, int ld, int row0,
                                             int kofs, int lane) {
  v16bf a;
  const int m   = row0 + (lane & 15);
  const int khi = (lane & 16) ? 16 : 0;
#pragma unroll
  for (int e = 0; e < 16; ++e) {
    const int k = kofs + e + khi;  // contiguous: lane half h -> k = 16h + e
    a[e] = p[(long long)m * ld + k];
  }
  return a;
}

// B operand from a buffer laid out [N][ld>=K] (K-contiguous per column):
// element (k, n) at p[n*ld + k]; per lane 16 contiguous bf16 (2x b128).
__device__ __forceinline__ v16bf load_bT_frag(const __bf16* p, int ld, int col0,
                                              int kofs, int lane) {
  v16bf bb;
  const int nn   = col0 + (lane & 15);
  const int koff = kofs + ((lane & 16) ? 16 : 0);
#pragma unroll
  for (int e = 0; e < 16; ++e) {
    bb[e] = p[(long long)nn * ld + koff + e];
  }
  return bb;
}

// ---------------------------------------------------------------------------
// Batched bf16 GEMM: C[M,N](f32) = A[M,K] @ B[K,N]  (+C if ACCUM)
// Requires: M % BM == 0, N % BN == 0, K % 32 == 0  (no bounds checks).
// 8 waves of 32x32 each (BM/32 x BN/32 wave grid), BK=32.
// A staged row-major [BM][40] (TDM with LDS padding: 4 DW pad / 16 DW);
// B staged K-contiguous [BN][40].
// ---------------------------------------------------------------------------
template <int BM, int BN, int ACCUM>
__global__ __launch_bounds__(256) void gemm_bf16_kernel(
    const __bf16* __restrict__ A, const __bf16* __restrict__ B,
    float* __restrict__ C, int K, int lda, int ldb, int ldc,
    long long sA, long long sB, long long sC, int bmodB) {
  __shared__ __bf16 At[BM][40];  // [m][k] +8 pad
  __shared__ __bf16 Bt[BN][40];  // [n][k] +8 pad

  constexpr int WROWS = BM / 32;  // waves along M
  constexpr int WCOLS = BN / 32;  // waves along N
  static_assert(WROWS * WCOLS == 8, "8 waves");

  const int tid = threadIdx.x, lane = tid & 31, wid = tid >> 5;
  const int wm = wid % WROWS, wn = wid / WROWS;
  const int bz = blockIdx.z;

  const __bf16* Ab = A + (long long)bz * sA;
  const __bf16* Bb = B + (long long)(bmodB ? (bz % bmodB) : bz) * sB;
  float*        Cb = C + (long long)bz * sC;

  const int m0 = blockIdx.y * BM;
  const int n0 = blockIdx.x * BN;

  v8f acc[2][2];
#pragma unroll
  for (int mi = 0; mi < 2; ++mi)
#pragma unroll
    for (int ni = 0; ni < 2; ++ni) {
      if (ACCUM) {
#pragma unroll
        for (int g = 0; g < 8; ++g) {
          const int row = m0 + wm * 32 + mi * 16 + g + ((lane & 16) ? 8 : 0);
          const int col = n0 + wn * 32 + ni * 16 + (lane & 15);
          acc[mi][ni][g] = Cb[(long long)row * ldc + col];
        }
      } else {
#pragma unroll
        for (int g = 0; g < 8; ++g) acc[mi][ni][g] = 0.f;
      }
    }

#if USE_TDM
  // LDS byte offset of At (ptrtoint of an addrspace(3) pointer).
  const unsigned lds_at =
      (unsigned)(unsigned long long)(__attribute__((address_space(3)))
                                         __bf16*)&At[0][0];
#endif

  constexpr int BCH = BN * 4;   // B chunks of 8 bf16
  constexpr int BCPR = BN / 8;  // B chunks per K-row
#if !USE_TDM
  constexpr int ACH = BM * 4;   // A chunks of 8 bf16
#endif

  for (int kk = 0; kk < K; kk += 32) {
#if USE_TDM
    // A: TDM 2D tile load, 32 (k) x BM (m) bf16, row stride lda elements,
    // LDS padding 4 DWORDs after every 16 DWORDs -> matches [BM][40].
    if (wid == 0) {
      const unsigned long long ga =
          (unsigned long long)(Ab + (long long)m0 * lda + kk);
      tdm_v4u g0;
      g0[0] = 1u;                                   // count=1 (valid)
      g0[1] = lds_at;                               // lds_addr
      g0[2] = (unsigned)ga;                         // global_addr[31:0]
      g0[3] = (unsigned)((ga >> 32) & 0x01FFFFFFu)  // global_addr[56:32]
              | (2u << 30);                         // type=2 (image)
      tdm_v8i g1;
      g1[0] = (int)((1u << 16)      // data_size = 2 bytes
                    | (1u << 20)    // pad_enable
                    | (3u << 22)    // pad_interval: 16 DW
                    | (3u << 25));  // pad_amount:   4 DW
      g1[1] = (int)(32u << 16);     // tensor_dim0[15:0] = 32 (in bits 63:48)
      g1[2] = (int)((unsigned)BM << 16);  // tensor_dim1[15:0] (bits 95:80)
      g1[3] = (int)(32u << 16);     // tile_dim0 = 32 (bits 127:112)
      g1[4] = BM;                   // tile_dim1 = BM (bits 143:128)
      g1[5] = lda;                  // tensor_dim0_stride[31:0]
      g1[6] = 0;
      g1[7] = 0;
      tdm_v4i gz4;
      gz4[0] = 0; gz4[1] = 0; gz4[2] = 0; gz4[3] = 0;
      tdm_v8i gz8;
#pragma unroll
      for (int i = 0; i < 8; ++i) gz8[i] = 0;
      __builtin_amdgcn_tensor_load_to_lds(g0, g1, gz4, gz4, gz8, 0);
    }
#else
    // A: vector global load -> vector LDS store (contiguous rows)
#pragma unroll
    for (int it = 0; it < ACH / 256; ++it) {
      const int e = tid + it * 256;
      const int r = e >> 2, c0 = (e & 3) * 8;
      v8bf av = *(const v8bf*)(Ab + (long long)(m0 + r) * lda + kk + c0);
      *(v8bf*)(&At[r][c0]) = av;
    }
#endif
    // B: vector global load -> K-transposed LDS store
#pragma unroll
    for (int it = 0; it < BCH / 256; ++it) {
      const int e = tid + it * 256;
      const int r = e / BCPR, c0 = (e % BCPR) * 8;
      v8bf bv = *(const v8bf*)(Bb + (long long)(kk + r) * ldb + n0 + c0);
#pragma unroll
      for (int i = 0; i < 8; ++i) Bt[c0 + i][r] = bv[i];
    }
#if USE_TDM
    if (wid == 0) __builtin_amdgcn_s_wait_tensorcnt(0);
#endif
    __syncthreads();

    v16bf af[2], bfv[2];
    af[0]  = load_a_frag(&At[0][0], 40, wm * 32 + 0, 0, lane);
    af[1]  = load_a_frag(&At[0][0], 40, wm * 32 + 16, 0, lane);
    bfv[0] = load_bT_frag(&Bt[0][0], 40, wn * 32 + 0, 0, lane);
    bfv[1] = load_bT_frag(&Bt[0][0], 40, wn * 32 + 16, 0, lane);
#pragma unroll
    for (int mi = 0; mi < 2; ++mi)
#pragma unroll
      for (int ni = 0; ni < 2; ++ni)
        acc[mi][ni] = wmma_bf16(af[mi], bfv[ni], acc[mi][ni]);
    __syncthreads();
  }

#pragma unroll
  for (int mi = 0; mi < 2; ++mi)
#pragma unroll
    for (int ni = 0; ni < 2; ++ni)
#pragma unroll
      for (int g = 0; g < 8; ++g) {
        const int row = m0 + wm * 32 + mi * 16 + g + ((lane & 16) ? 8 : 0);
        const int col = n0 + wn * 32 + ni * 16 + (lane & 15);
        Cb[(long long)row * ldc + col] = acc[mi][ni][g];
      }
}

// ---------------------------------------------------------------------------
// Vectorized f32 -> bf16 convert (n % 4 == 0).
// ---------------------------------------------------------------------------
__global__ void f32_to_bf16_kernel(const float* __restrict__ in,
                                   __bf16* __restrict__ out, long long n4) {
  const long long t = (long long)blockIdx.x * 256 + threadIdx.x;
  if (t < n4) {
    v4f x = *(const v4f*)(in + t * 4);
    v4bf y;
#pragma unroll
    for (int i = 0; i < 4; ++i) y[i] = (__bf16)x[i];
    *(v4bf*)(out + t * 4) = y;
  }
}

// ---------------------------------------------------------------------------
// Split heads -> bf16 q (+biases, [B,NH,Q,DH]), k ([B,NH,K,DH]), v ([B,NH,K,DH])
// heads: f32 [K*B, 3*NH*DH], row index = j*B + b.  (k kept key-major so score
// GEMM B-fragments are contiguous 16-element runs.)
// ---------------------------------------------------------------------------
__global__ void prep_qkv_kernel(const float* __restrict__ heads,
                                const float* __restrict__ rwb,
                                const float* __restrict__ rrb,
                                __bf16* __restrict__ qwb, __bf16* __restrict__ qrb,
                                __bf16* __restrict__ kb, __bf16* __restrict__ vb) {
  const long long t = (long long)blockIdx.x * 256 + threadIdx.x;
  if (t >= (long long)KLEN * BD * NHD * DHD) return;
  const int d = (int)(t & 63);
  const int n = (int)((t >> 6) & 15);
  const int b = (int)((t >> 10) & 7);
  const int j = (int)(t >> 13);
  const int col = n * 64 + d;
  const float* hrow = heads + (long long)(j * BD + b) * HD3;
  kb[((long long)(b * NHD + n) * KLEN + j) * DHD + d] = (__bf16)hrow[HD + col];
  vb[((long long)(b * NHD + n) * KLEN + j) * DHD + d] = (__bf16)hrow[2 * HD + col];
  if (j >= MLEND) {
    const int i = j - MLEND;
    const float qv = hrow[col];
    qwb[((long long)(b * NHD + n) * QD + i) * DHD + d] = (__bf16)(qv + rwb[col]);
    qrb[((long long)(b * NHD + n) * QD + i) * DHD + d] = (__bf16)(qv + rrb[col]);
  }
}

// rk f32 [K, NH*DH] -> bf16 [NH, K, DH] (key-major, matches kb layout)
__global__ void prep_rk_kernel(const float* __restrict__ rk,
                               __bf16* __restrict__ rkb) {
  const long long t = (long long)blockIdx.x * 256 + threadIdx.x;
  if (t >= (long long)NHD * KLEN * DHD) return;
  const int d = (int)(t & 63);
  const int j = (int)((t >> 6) & 1023);
  const int n = (int)(t >> 16);
  rkb[t] = (__bf16)rk[(long long)j * HD + n * 64 + d];
}

// ---------------------------------------------------------------------------
// Fused score kernel: per (b, n, 16-query tile):
//   S = (q+r_w_bias) k^T  +  rel_shift((q+r_r_bias) rk^T), masked band softmax,
//   writes prob (bf16, zeros outside the band  j in (qi, qi+512]).
// Scores live only in LDS (16 x 1024 f32 = 64KB).  4 waves, 128 threads.
// B fragments come straight from global [K,DH] rows (2x b128 per fragment).
// ---------------------------------------------------------------------------
__global__ __launch_bounds__(128) void score_softmax_kernel(
    const __bf16* __restrict__ qwb, const __bf16* __restrict__ qrb,
    const __bf16* __restrict__ kb, const __bf16* __restrict__ rkb,
    __bf16* __restrict__ prob) {
  __shared__ float sS[16][1024];

  const int q0 = blockIdx.x * 16;
  const int n  = blockIdx.y;
  const int b  = blockIdx.z;
  const int tid = threadIdx.x, lane = tid & 31, wv = tid >> 5;

  const __bf16* qw = qwb + ((long long)(b * NHD + n) * QD + q0) * DHD;
  const __bf16* qr = qrb + ((long long)(b * NHD + n) * QD + q0) * DHD;
  const __bf16* kt = kb + (long long)(b * NHD + n) * KLEN * DHD;
  const __bf16* rk = rkb + (long long)n * KLEN * DHD;

  v16bf aw[2], ar[2];
#pragma unroll
  for (int ks = 0; ks < 2; ++ks) {
    aw[ks] = load_a_frag(qw, DHD, 0, ks * 32, lane);
    ar[ks] = load_a_frag(qr, DHD, 0, ks * 32, lane);
  }

  // Phase A: AC tile -> sS (each wave owns 256 contiguous columns)
  for (int t = 0; t < 16; ++t) {
    const int c0 = wv * 256 + t * 16;
    v8f acc;
#pragma unroll
    for (int g = 0; g < 8; ++g) acc[g] = 0.f;
#pragma unroll
    for (int ks = 0; ks < 2; ++ks) {
      v16bf bfv = load_bT_frag(kt, DHD, c0, ks * 32, lane);
      acc = wmma_bf16(aw[ks], bfv, acc);
    }
#pragma unroll
    for (int g = 0; g < 8; ++g) {
      const int row = g + ((lane & 16) ? 8 : 0);
      sS[row][c0 + (lane & 15)] = acc[g];
    }
  }
  __syncthreads();

  // Phase B: BD' tile, scatter-added with the TXL rel-shift:
  //   BD[i][j] = BD'[i][u],  u = j + 511 - i  =>  j = u + i - 511.
  for (int t = 0; t < 16; ++t) {
    const int c0 = wv * 256 + t * 16;
    v8f acc;
#pragma unroll
    for (int g = 0; g < 8; ++g) acc[g] = 0.f;
#pragma unroll
    for (int ks = 0; ks < 2; ++ks) {
      v16bf bfv = load_bT_frag(rk, DHD, c0, ks * 32, lane);
      acc = wmma_bf16(ar[ks], bfv, acc);
    }
#pragma unroll
    for (int g = 0; g < 8; ++g) {
      const int row = g + ((lane & 16) ? 8 : 0);
      const int u = c0 + (lane & 15);
      const int j = u + (q0 + row) - 511;
      if (j >= 0 && j < KLEN) sS[row][j] += acc[g];
    }
  }
  __syncthreads();

  // Softmax over the valid band j in [qi+1, qi+512]; one wave per 4 rows.
  const float scale = 0.125f;  // 1/sqrt(64)
  for (int rr = 0; rr < 4; ++rr) {
    const int r = wv * 4 + rr;
    const int qi = q0 + r;
    float vals[16];
    float mx = -3.4e38f;
#pragma unroll
    for (int t = 0; t < 16; ++t) {
      const int j = qi + 1 + lane + t * 32;
      const float s = sS[r][j] * scale;
      vals[t] = s;
      mx = fmaxf(mx, s);
    }
    for (int o = 16; o > 0; o >>= 1) mx = fmaxf(mx, __shfl_xor(mx, o, 32));
    float sum = 0.f;
#pragma unroll
    for (int t = 0; t < 16; ++t) {
      const float p = __expf(vals[t] - mx);
      sum += p;
      sS[r][qi + 1 + lane + t * 32] = p;
    }
    for (int o = 16; o > 0; o >>= 1) sum += __shfl_xor(sum, o, 32);
    const float inv = 1.f / sum;
    __bf16* pr = prob + ((long long)(b * NHD + n) * QD + qi) * KLEN;
    // Each lane owns 32 *contiguous* columns -> wide global stores.
#pragma unroll
    for (int t = 0; t < 32; ++t) {
      const int j = lane * 32 + t;
      const bool inb = (j > qi) && (j <= qi + 512);
      pr[j] = (__bf16)(inb ? sS[r][j] * inv : 0.f);
    }
  }
}

// rel_prob gather: relp[b,n,i,s] = prob[b,n,i, rel_pos[i,s,b,n]]
__global__ void gather_rel_kernel(const __bf16* __restrict__ prob,
                                  const int* __restrict__ rel_pos,
                                  __bf16* __restrict__ relp) {
  const long long t = (long long)blockIdx.x * 256 + threadIdx.x;
  if (t >= (long long)BD * NHD * QD * SPAND) return;
  const int s = (int)(t & 511);
  const int i = (int)((t >> 9) & 511);
  const int n = (int)((t >> 18) & 15);
  const int b = (int)(t >> 22);
  const int rp = rel_pos[(((long long)i * SPAND + s) * BD + b) * NHD + n];
  relp[t] = prob[((long long)(b * NHD + n) * QD + i) * KLEN + rp];
}

// attn_vec f32 [B,NH,Q,DH] -> bf16 [Q*B, NH*DH]  (reference reshape order)
__global__ void conv_av_kernel(const float* __restrict__ attnvec,
                               __bf16* __restrict__ avb) {
  const long long t = (long long)blockIdx.x * 256 + threadIdx.x;
  if (t >= (long long)QD * BD * HD) return;
  const int d = (int)(t & 63);
  const int n = (int)((t >> 6) & 15);
  const int b = (int)((t >> 10) & 7);
  const int q = (int)(t >> 13);
  avb[t] = (__bf16)attnvec[((long long)(b * NHD + n) * QD + q) * DHD + d];
}

// out = LayerNorm(w + attn_out) * gamma + beta, one block per row of D=1024
__global__ __launch_bounds__(256) void add_ln_kernel(
    const float* __restrict__ w, const float* __restrict__ ao,
    const float* __restrict__ gamma, const float* __restrict__ beta,
    float* __restrict__ out) {
  __shared__ float s1[256];
  __shared__ float s2[256];
  const int row = blockIdx.x, tid = threadIdx.x;
  float x[4];
  float sum = 0.f, sq = 0.f;
#pragma unroll
  for (int t = 0; t < 4; ++t) {
    const int d = t * 256 + tid;
    const float v = w[(long long)row * DD + d] + ao[(long long)row * DD + d];
    x[t] = v;
    sum += v;
    sq += v * v;
  }
  s1[tid] = sum;
  s2[tid] = sq;
  __syncthreads();
  for (int o = 128; o > 0; o >>= 1) {
    if (tid < o) { s1[tid] += s1[tid + o]; s2[tid] += s2[tid + o]; }
    __syncthreads();
  }
  const float mu  = s1[0] * (1.f / DD);
  const float var = s2[0] * (1.f / DD) - mu * mu;
  const float inv = rsqrtf(var + 1e-5f);
#pragma unroll
  for (int t = 0; t < 4; ++t) {
    const int d = t * 256 + tid;
    out[(long long)row * DD + d] = (x[t] - mu) * inv * gamma[d] + beta[d];
  }
}

// ---------------------------------------------------------------------------
// Host-side orchestration.
// ---------------------------------------------------------------------------
extern "C" void kernel_launch(void* const* d_in, const int* in_sizes, int n_in,
                              void* d_out, int out_size, void* d_ws,
                              size_t ws_size, hipStream_t stream) {
  (void)in_sizes; (void)n_in; (void)out_size; (void)ws_size;

  const float* w     = (const float*)d_in[0];
  const float* r     = (const float*)d_in[1];
  const float* mems  = (const float*)d_in[2];
  const float* W_qkv = (const float*)d_in[3];
  const float* W_r   = (const float*)d_in[4];
  const float* W_o   = (const float*)d_in[5];
  const float* rwb   = (const float*)d_in[6];
  const float* rrb   = (const float*)d_in[7];
  const float* v_rpe = (const float*)d_in[8];
  const float* gamma = (const float*)d_in[9];
  const float* beta  = (const float*)d_in[10];
  // d_in[11] = attn_mask (bool) -- analytic band j in (i, i+512], unused
  const int* rel_pos = (const int*)d_in[12];
  float* out = (float*)d_out;

  // Workspace carve (bytes, 256-aligned)
  size_t off = 0;
  char* base = (char*)d_ws;
  auto carve = [&](size_t bytes) -> char* {
    char* p = base + off;
    off += (bytes + 255) & ~(size_t)255;
    return p;
  };
  __bf16* catb   = (__bf16*)carve((size_t)KLEN * BD * DD * 2);   // 16 MB
  __bf16* wqkvb  = (__bf16*)carve((size_t)DD * HD3 * 2);
  __bf16* rb     = (__bf16*)carve((size_t)KLEN * DD * 2);
  __bf16* wrb    = (__bf16*)carve((size_t)DD * HD * 2);
  __bf16* wob    = (__bf16*)carve((size_t)HD * DD * 2);
  __bf16* vrpeb  = (__bf16*)carve((size_t)SPAND * HD * 2);
  float*  heads  = (float*)carve((size_t)KLEN * BD * HD3 * 4);   // 96 MB
  float*  rkf    = (float*)carve((size_t)KLEN * HD * 4);
  __bf16* qwb    = (__bf16*)carve((size_t)BD * NHD * QD * DHD * 2);
  __bf16* qrb    = (__bf16*)carve((size_t)BD * NHD * QD * DHD * 2);
  __bf16* kbb    = (__bf16*)carve((size_t)BD * NHD * KLEN * DHD * 2);
  __bf16* vb     = (__bf16*)carve((size_t)BD * NHD * KLEN * DHD * 2);
  __bf16* rkb    = (__bf16*)carve((size_t)NHD * KLEN * DHD * 2);
  __bf16* prob   = (__bf16*)carve((size_t)BD * NHD * QD * KLEN * 2);  // 134 MB
  float*  attnv  = (float*)carve((size_t)BD * NHD * QD * DHD * 4);
  __bf16* relpb  = (__bf16*)carve((size_t)BD * NHD * QD * SPAND * 2); // 67 MB
  __bf16* avb    = (__bf16*)carve((size_t)QD * BD * HD * 2);
  float*  attnout= (float*)carve((size_t)QD * BD * DD * 4);

  const dim3 b256(256);
  auto conv = [&](const float* src, __bf16* dst, long long n) {
    const long long n4 = n / 4;
    f32_to_bf16_kernel<<<dim3((unsigned)((n4 + 255) / 256)), b256, 0, stream>>>(
        src, dst, n4);
  };

  // 1) bf16 conversions; cat = [mems; w]
  conv(mems, catb, (long long)MLEND * BD * DD);
  conv(w, catb + (long long)MLEND * BD * DD, (long long)QD * BD * DD);
  conv(W_qkv, wqkvb, (long long)DD * HD3);
  conv(r, rb, (long long)KLEN * DD);
  conv(W_r, wrb, (long long)DD * HD);
  conv(W_o, wob, (long long)HD * DD);
  conv(v_rpe, vrpeb, (long long)SPAND * HD);

  // 2) heads = cat @ W_qkv   [8192 x 1024] x [1024 x 3072]   (tile 64x128)
  gemm_bf16_kernel<64, 128, 0><<<dim3(HD3 / 128, (KLEN * BD) / 64, 1), b256, 0,
                                 stream>>>(catb, wqkvb, heads, DD, DD, HD3, HD3,
                                           0, 0, 0, 0);
  // 3) rk = r @ W_r          [1024 x 1024] x [1024 x 1024]   (tile 64x128)
  gemm_bf16_kernel<64, 128, 0><<<dim3(HD / 128, KLEN / 64, 1), b256, 0,
                                 stream>>>(rb, wrb, rkf, DD, DD, HD, HD, 0, 0,
                                           0, 0);

  // 4) split q/k/v (+biases), rk reshape — all key-major bf16
  prep_qkv_kernel<<<dim3((unsigned)(((long long)KLEN * BD * NHD * DHD + 255) / 256)),
                    b256, 0, stream>>>(heads, rwb, rrb, qwb, qrb, kbb, vb);
  prep_rk_kernel<<<dim3((unsigned)(((long long)NHD * KLEN * DHD + 255) / 256)),
                   b256, 0, stream>>>(rkf, rkb);

  // 5) fused AC + rel-shifted BD + mask + softmax -> prob (bf16)
  score_softmax_kernel<<<dim3(QD / 16, NHD, BD), dim3(128), 0, stream>>>(
      qwb, qrb, kbb, rkb, prob);

  // 6) attn_vec = prob @ v   (128 batches of [512x1024]x[1024x64], tile 128x64)
  gemm_bf16_kernel<128, 64, 0><<<dim3(DHD / 64, QD / 128, BD * NHD), b256, 0,
                                 stream>>>(prob, vb, attnv, KLEN, KLEN, DHD,
                                           DHD, (long long)QD * KLEN,
                                           (long long)KLEN * DHD,
                                           (long long)QD * DHD, 0);

  // 7) rel_prob gather, then attn_vec += rel_prob @ v_rpe (accumulating GEMM)
  gather_rel_kernel<<<dim3((unsigned)(((long long)BD * NHD * QD * SPAND + 255) / 256)),
                      b256, 0, stream>>>(prob, rel_pos, relpb);
  gemm_bf16_kernel<128, 64, 1><<<dim3(DHD / 64, QD / 128, BD * NHD), b256, 0,
                                 stream>>>(relpb, vrpeb, attnv, SPAND, SPAND,
                                           HD, DHD, (long long)QD * SPAND,
                                           (long long)DHD,
                                           (long long)QD * DHD, /*bmodB=*/NHD);

  // 8) attn_out = attn_vec.reshape(Q,B,NH*DH) @ W_o   (tile 64x128)
  conv_av_kernel<<<dim3((unsigned)(((long long)QD * BD * HD + 255) / 256)), b256,
                   0, stream>>>(attnv, avb);
  gemm_bf16_kernel<64, 128, 0><<<dim3(DD / 128, (QD * BD) / 64, 1), b256, 0,
                                 stream>>>(avb, wob, attnout, HD, HD, DD, DD, 0,
                                           0, 0, 0);

  // 9) out = LayerNorm(w + attn_out)
  add_ln_kernel<<<dim3(QD * BD), b256, 0, stream>>>(w, attnout, gamma, beta, out);
}
